// NeuralRDE_58377195487357
// MI455X (gfx1250) — matrix operand order repaired and checked by reading, not compiled, not run on hardware
//
#include <hip/hip_runtime.h>

typedef _Float16 f16;
typedef __attribute__((ext_vector_type(16))) _Float16 v16h;
typedef __attribute__((ext_vector_type(8)))  _Float16 v8h;
typedef __attribute__((ext_vector_type(4)))  _Float16 v4h;
typedef __attribute__((ext_vector_type(8)))  float    v8f;

namespace {
constexpr int B = 512, T = 1000, C = 8, IN = 8, H = 64, MLPW = 128;
constexpr int CH = C * H;          // 512
constexpr int BT = 16;             // batch rows per block (WMMA M)
constexpr int NTHREADS = 512;      // 16 waves (wave32)

// LDS layout (bytes); all chunks 32B-aligned.
constexpr int OFF_W2  = 0;                       // 512*128 f16 = 131072
constexpr int OFF_W1  = OFF_W2 + CH * MLPW * 2;  // 128*64  f16 = 16384
constexpr int OFF_Z16 = OFF_W1 + MLPW * H * 2;   // 16*64   f16 = 2048
constexpr int OFF_H16 = OFF_Z16 + BT * H * 2;    // 16*128  f16 = 4096
constexpr int OFF_ZT  = OFF_H16 + BT * MLPW * 2; // 16*64   f32 = 4096 (init tmp)
constexpr int OFF_B2  = OFF_ZT + BT * H * 4;     // 512     f32 = 2048
constexpr int OFF_B1  = OFF_B2 + CH * 4;         // 128     f32 = 512
constexpr int OFF_DX  = OFF_B1 + MLPW * 4;       // 16*8    f32 = 512
constexpr int OFF_WR  = OFF_DX + BT * C * 4;     // 64      f32 = 256
constexpr int OFF_OA  = OFF_WR + H * 4;          // 16      f32 = 64  (readout acc)
constexpr int SMEM_BYTES = OFF_OA + BT * 4;      // ~157 KB of 320 KB WGP LDS
}

__device__ __forceinline__ v8f wmma32(v16h a, v16h b, v8f c) {
  // v_wmma_f32_16x16x32_f16  D = A*B + C
  return __builtin_amdgcn_wmma_f32_16x16x32_f16(
      /*neg_a=*/false, a, /*neg_b=*/false, b,
      /*c_mod=*/(short)0, c, /*reuse_a=*/false, /*reuse_b=*/false);
}

__device__ __forceinline__ float fast_tanh(float x) {
#if __has_builtin(__builtin_amdgcn_tanhf)
  return __builtin_amdgcn_tanhf(x);     // gfx1250 v_tanh_f32
#elif __has_builtin(__builtin_amdgcn_tanh_f32)
  return __builtin_amdgcn_tanh_f32(x);
#else
  return tanhf(x);
#endif
}

// A-matrix 16x32 f16 fragment: lanes 0-15 hold row m=lane, K = k0+{0..15};
// lanes 16-31 hold row m=lane-16, K = k0+{16..31}.
__device__ __forceinline__ v16h ldA(const f16* row, int k0, int lane) {
  return *(const v16h*)(row + k0 + ((lane >> 4) << 4));
}

// B-matrix 32x16 f16 fragment: lane holds column n=lane&15; lanes 0-15 carry
// K=k0..k0+15, lanes 16-31 carry K=k0+16..k0+31 (contiguous 32B per lane).
__device__ __forceinline__ v16h ldB(const f16* row, int k0, int lane) {
  return *(const v16h*)(row + k0 + ((lane >> 4) << 4));
}

__global__ __launch_bounds__(NTHREADS)
void cde_scan_kernel(const float* __restrict__ u0,
                     const float* __restrict__ coeffs,
                     const float* __restrict__ W1, const float* __restrict__ b1,
                     const float* __restrict__ W2, const float* __restrict__ b2,
                     const float* __restrict__ Wi, const float* __restrict__ bi,
                     const float* __restrict__ Wr, const float* __restrict__ br,
                     float* __restrict__ out) {
  extern __shared__ __align__(32) char smem[];
  f16*   W2h = (f16*)(smem + OFF_W2);
  f16*   W1h = (f16*)(smem + OFF_W1);
  f16*   z16 = (f16*)(smem + OFF_Z16);
  f16*   h16 = (f16*)(smem + OFF_H16);
  float* zt  = (float*)(smem + OFF_ZT);
  float* b2l = (float*)(smem + OFF_B2);
  float* b1l = (float*)(smem + OFF_B1);
  float* dxl = (float*)(smem + OFF_DX);
  float* wrl = (float*)(smem + OFF_WR);
  float* oacc = (float*)(smem + OFF_OA);

  const int tid  = threadIdx.x;
  const int wave = tid >> 5;
  const int lane = tid & 31;
  const int mrow = lane & 15;
  const int mb   = (lane >> 4) << 3;   // D rows: lanes<16 -> M=r, else M=r+8
  const int hbit = (lane >> 3) & 1;    // column sub-group within the 16-col tile
  const int b0   = blockIdx.x * BT;
  const float br0 = br[0];

  // ---- Stage weights fp32 -> fp16 into LDS once (reused 999x) ----
  for (int i = tid * 4; i < CH * MLPW; i += NTHREADS * 4) {
    float4 w = *(const float4*)(W2 + i);
    *(v4h*)(W2h + i) = v4h{(f16)w.x, (f16)w.y, (f16)w.z, (f16)w.w};
  }
  for (int i = tid * 4; i < MLPW * H; i += NTHREADS * 4) {
    float4 w = *(const float4*)(W1 + i);
    *(v4h*)(W1h + i) = v4h{(f16)w.x, (f16)w.y, (f16)w.z, (f16)w.w};
  }
  for (int i = tid; i < CH;   i += NTHREADS) b2l[i] = b2[i];
  for (int i = tid; i < MLPW; i += NTHREADS) b1l[i] = b1[i];
  for (int i = tid; i < H;    i += NTHREADS) wrl[i] = Wr[i];

  // ---- z0 = u0 @ Wi^T + bi ----
  for (int i = tid; i < BT * H; i += NTHREADS) {
    int m = i >> 6, hc = i & 63;
    float s = bi[hc];
    const float* up = u0 + (size_t)(b0 + m) * IN;
    #pragma unroll
    for (int j = 0; j < IN; ++j) s += up[j] * Wi[hc * IN + j];
    zt[i] = s;
    z16[i] = (f16)s;
  }
  __syncthreads();

  // ---- seed readout accumulator: oacc[m] = dot(z0[m], Wr) ----
  {
    int m = tid >> 5, seg = tid & 31;
    float s = 0.f;
    #pragma unroll
    for (int j = 0; j < 2; ++j) {
      int hc = seg * 2 + j;
      s += zt[m * H + hc] * wrl[hc];
    }
    s += __shfl_xor(s, 1); s += __shfl_xor(s, 2); s += __shfl_xor(s, 4);
    s += __shfl_xor(s, 8); s += __shfl_xor(s, 16);
    if (seg == 0) oacc[m] = s;
  }

  // ---- Preload constant B-fragments into VGPRs for the whole scan ----
  v16h bf1[2];
  {
    const f16* row = W1h + ((wave & 7) * 16 + mrow) * H;
    bf1[0] = ldB(row, 0,  lane);
    bf1[1] = ldB(row, 32, lane);
  }
  v16h wf2[2][4];
  #pragma unroll
  for (int j = 0; j < 2; ++j) {
    const f16* row = W2h + ((wave * 2 + j) * 16 + mrow) * MLPW;
    #pragma unroll
    for (int ks = 0; ks < 4; ++ks) wf2[j][ks] = ldB(row, ks * 32, lane);
  }

  // ---- Loop-invariant per-lane scalars ----
  const int   hh0    = wave * 4 + hbit;       // owned state columns (tile j=0)
  const int   hh1    = wave * 4 + 2 + hbit;   // and tile j=1
  const float wr0    = wrl[hh0];
  const float wr1    = wrl[hh1];
  const float bias1  = b1l[(wave & 7) * 16 + mrow];
  const float bias20 = b2l[(wave * 2) * 16 + mrow];
  const float bias21 = b2l[(wave * 2 + 1) * 16 + mrow];

  // ---- f32 state lives in registers: group-owned (m, hh) pairs ----
  float zr0[8], zr1[8];
  #pragma unroll
  for (int r = 0; r < 8; ++r) {
    zr0[r] = zt[(mb + r) * H + hh0];
    zr1[r] = zt[(mb + r) * H + hh1];
  }

  // ---- Software-pipelined dx loads (one element per thread, tid<128) ----
  float cprev = 0.f, ccur = 0.f;
  const float* cptr = coeffs;
  if (tid < BT * C) {
    int m = tid >> 3, cc = tid & 7;
    cptr = coeffs + (size_t)(b0 + m) * T * C + cc;
    cprev = cptr[0];   // coeffs[:,0]
    ccur  = cptr[C];   // coeffs[:,1]
  }

  for (int t = 1; t < T; ++t) {
    if (tid < BT * C) {
      dxl[tid] = ccur - cprev;                      // dX[t-1]
      cprev = ccur;
      if (t + 1 < T) ccur = cptr[(size_t)(t + 1) * C];  // in flight all step
    }
    __syncthreads();  // B1: dxl/z16/oacc(t-1) ready

    // drain readout of step t-1, reset accumulator (ordered vs adds by B2/B1)
    if (tid < BT) {
      out[(size_t)(b0 + tid) * T + (t - 1)] = oacc[tid] + br0;
      oacc[tid] = 0.f;
    }

    float dxr[8];
    {
      int ccl = lane & 7;
      #pragma unroll
      for (int r = 0; r < 8; ++r) dxr[r] = dxl[(mb + r) * C + ccl];
    }

    // ---- GEMM1: h = relu(z @ W1^T + b1); waves 0..7 -> one 16-col tile ----
    if (wave < 8) {
      v16h a0 = ldA(z16 + mrow * H, 0,  lane);
      v16h a1 = ldA(z16 + mrow * H, 32, lane);
      v8f acc = {};
      acc = wmma32(a0, bf1[0], acc);
      acc = wmma32(a1, bf1[1], acc);
      int ncol = wave * 16 + mrow;
      #pragma unroll
      for (int r = 0; r < 8; ++r) {
        float v = acc[r] + bias1;
        h16[(mb + r) * MLPW + ncol] = (f16)(v > 0.f ? v : 0.f);
      }
    }
    __syncthreads();  // B2: h16 complete

    // ---- GEMM2: f = tanh(h @ W2^T + b2); fused einsum + state + readout ----
    {
      v16h ha[4];
      #pragma unroll
      for (int ks = 0; ks < 4; ++ks) ha[ks] = ldA(h16 + mrow * MLPW, ks * 32, lane);
      v8f acc0 = {}, acc1 = {};
      #pragma unroll
      for (int ks = 0; ks < 4; ++ks) acc0 = wmma32(ha[ks], wf2[0][ks], acc0);
      #pragma unroll
      for (int ks = 0; ks < 4; ++ks) acc1 = wmma32(ha[ks], wf2[1][ks], acc1);
      #pragma unroll
      for (int r = 0; r < 8; ++r) {
        float p0 = fast_tanh(acc0[r] + bias20) * dxr[r];
        float p1 = fast_tanh(acc1[r] + bias21) * dxr[r];
        // reduce over c: 8 consecutive lanes == 8 consecutive f columns
        p0 += __shfl_xor(p0, 1); p0 += __shfl_xor(p0, 2); p0 += __shfl_xor(p0, 4);
        p1 += __shfl_xor(p1, 1); p1 += __shfl_xor(p1, 2); p1 += __shfl_xor(p1, 4);
        zr0[r] += p0;   // group-uniform after reduction
        zr1[r] += p1;
      }
      if ((lane & 7) == 0) {   // one writer per 8-lane group
        #pragma unroll
        for (int r = 0; r < 8; ++r) {
          z16[(mb + r) * H + hh0] = (f16)zr0[r];
          z16[(mb + r) * H + hh1] = (f16)zr1[r];
          atomicAdd(&oacc[mb + r], zr0[r] * wr0 + zr1[r] * wr1);
        }
      }
    }
    // no barrier here: B1 of step t+1 orders z16/oacc writes vs consumers
  }

  __syncthreads();
  if (tid < BT) out[(size_t)(b0 + tid) * T + (T - 1)] = oacc[tid] + br0;
}

extern "C" void kernel_launch(void* const* d_in, const int* in_sizes, int n_in,
                              void* d_out, int out_size, void* d_ws, size_t ws_size,
                              hipStream_t stream) {
  (void)in_sizes; (void)n_in; (void)out_size; (void)d_ws; (void)ws_size;
  const float* u0 = (const float*)d_in[0];
  const float* co = (const float*)d_in[1];
  const float* W1 = (const float*)d_in[2];
  const float* b1 = (const float*)d_in[3];
  const float* W2 = (const float*)d_in[4];
  const float* b2 = (const float*)d_in[5];
  const float* Wi = (const float*)d_in[6];
  const float* bi = (const float*)d_in[7];
  const float* Wr = (const float*)d_in[8];
  const float* br = (const float*)d_in[9];
  float* out = (float*)d_out;

  dim3 grid(B / BT);     // 32 blocks, one 16-row batch tile each
  dim3 block(NTHREADS);  // 16 wave32 waves
  hipLaunchKernelGGL(cde_scan_kernel, grid, block, SMEM_BYTES, stream,
                     u0, co, W1, b1, W2, b2, Wi, bi, Wr, br, out);
}
